// _Keypointer_13615046329059
// MI455X (gfx1250) — matrix operand order repaired and checked by reading, not compiled, not run on hardware
//
#include <hip/hip_runtime.h>

// Keypoint heatmap bicubic resize (56x56 -> 96x96) + argmax, per (roi, keypoint).
// Two f16 WMMA GEMMs (fp32 accumulate) per heatmap, then argmax and an exact
// fp32 4x4-tap rescore of the winning position.
//
// Block = 192 threads = 6 waves. GEMM1 is computed transposed
// (T^T = mask^T @ M^T) so each D-fragment stores to LDS with a single
// ds_store_b128 in exactly the row-major layout GEMM2's A-fragments read.
// Each GEMM issues its WMMAs as independent accumulator chains (kk pass 0 for
// all tiles, then kk pass 1) to avoid WMMA->WMMA hazard NOPs.

typedef __attribute__((ext_vector_type(16))) _Float16 v16h;
typedef __attribute__((ext_vector_type(8)))  _Float16 v8h;
typedef __attribute__((ext_vector_type(8)))  float    v8f;
typedef __attribute__((ext_vector_type(4)))  float    v4f;

#define RR   512
#define KPT  17
#define HIN  56     // input heatmap side
#define ROI  96     // output side
#define PKD  64     // padded inner dim (56 -> 64)
#define NTHR 192    // 6 waves

__device__ __forceinline__ float cubicw(float x) {
    const float A = -0.75f;
    float ax = fabsf(x);
    if (ax <= 1.0f) return ((A + 2.0f) * ax - (A + 3.0f)) * ax * ax + 1.0f;
    if (ax < 2.0f)  return A * (((ax - 5.0f) * ax + 8.0f) * ax - 4.0f);
    return 0.0f;
}

// Compute the <=4 border-replicated bicubic taps for output coord o.
// Writes accumulated weights into w[0..3] (relative to *i0), zero-filled.
__device__ __forceinline__ void bicubic_taps(int o, int* i0out, float w[4]) {
    const float scale = (float)HIN / (float)ROI;
    float src = ((float)o + 0.5f) * scale - 0.5f;
    int base = (int)floorf(src);
    int i0 = base - 1 < 0 ? 0 : base - 1;
    w[0] = w[1] = w[2] = w[3] = 0.0f;
#pragma unroll
    for (int d = 0; d < 4; ++d) {
        int t = base - 1 + d;
        float ww = cubicw(src - (float)t);
        int tc = t < 0 ? 0 : (t > HIN - 1 ? HIN - 1 : t);
        w[tc - i0] += ww;           // duplicates at borders accumulate
    }
    *i0out = i0;
}

__device__ __forceinline__ v16h load_afrag(const _Float16* base) {
    // 16-bit A fragment, interleaved K layout: two 16B LDS reads
    v8h a0 = *(const v8h*)base;
    v8h a1 = *(const v8h*)(base + 16);
    return __builtin_shufflevector(a0, a1, 0,1,2,3,4,5,6,7,8,9,10,11,12,13,14,15);
}

__global__ __launch_bounds__(NTHR)
void kp_heatmap_kernel(const float* __restrict__ masks,
                       const float* __restrict__ boxes,
                       float* __restrict__ out)
{
    // LDS: resize matrix (f16), transposed mask (f16), intermediate T (f16)
    __shared__ __align__(32) _Float16 sM[ROI * PKD];   // [96][64]  M rows (taps)
    __shared__ __align__(32) _Float16 sB[PKD * PKD];   // [64][64]  mask^T: sB[j][i] = mask[i][j]
    __shared__ __align__(32) _Float16 sT[ROI * PKD];   // [96][64]  T row-major: sT[m][j]
    __shared__ float sRv[6];
    __shared__ int   sRp[6];

    const int tid  = threadIdx.x;
    const int lane = tid & 31;
    const int wid  = tid >> 5;       // 0..5
    const int g    = lane >> 4;      // half-wave group
    const int ln   = lane & 15;
    const int rk   = blockIdx.x;     // r*17 + k
    const float* __restrict__ m = masks + (size_t)rk * (HIN * HIN);

    // ---- build f16 resize matrix rows (one thread per output row) ----
    if (tid < ROI) {
        int i0; float w[4];
        bicubic_taps(tid, &i0, w);
        _Float16* row = &sM[tid * PKD];
        v8h z = {};
#pragma unroll
        for (int c = 0; c < 8; ++c) *(v8h*)(row + c * 8) = z;   // 8x b128 zero
#pragma unroll
        for (int d = 0; d < 4; ++d) {
            int c = i0 + d;
            if (c < HIN) row[c] = (_Float16)w[d];
        }
    }
    // ---- zero padded transposed-mask tile ----
    for (int i = tid; i < (PKD * PKD) / 2; i += NTHR)
        ((unsigned int*)sB)[i] = 0u;
    __syncthreads();

    // ---- load mask as float4 (b128), f32 -> f16, transpose into sB[j][i] ----
    for (int t = tid; t < (HIN * HIN) / 4; t += NTHR) {   // 784 float4 chunks
        v4f v = *(const v4f*)(m + t * 4);
#pragma unroll
        for (int e = 0; e < 4; ++e) {
            int idx = t * 4 + e;
            int i = idx / HIN;
            int j = idx - i * HIN;
            sB[j * PKD + i] = (_Float16)v[e];
        }
    }
    __syncthreads();

    const v8f vzero = {};

    // ============ GEMM1 (transposed): T^T[64x96] = mask^T[64x64] @ M^T[64x96] ====
    // Wave `wid` owns N-tile mt = wid (roi-row block), loops all 4 M-tiles jt.
    // D frag: lane = fixed roi-row, VGPRs = 8 consecutive j -> one b128 store.
    {
        const int mt = wid;
        // B fragments (M^T), interleaved K layout: depend only on (mt, kk)
        v16h b0 = load_afrag(&sM[(mt * 16 + ln) * PKD + 0 * 32 + 8 * g]);
        v16h b1 = load_afrag(&sM[(mt * 16 + ln) * PKD + 1 * 32 + 8 * g]);
        v16h a0[4], a1[4];
#pragma unroll
        for (int jt = 0; jt < 4; ++jt) {
            a0[jt] = load_afrag(&sB[(jt * 16 + ln) * PKD + 0 * 32 + 8 * g]);
            a1[jt] = load_afrag(&sB[(jt * 16 + ln) * PKD + 1 * 32 + 8 * g]);
        }
        v8f acc[4];
#pragma unroll
        for (int jt = 0; jt < 4; ++jt)
            acc[jt] = __builtin_amdgcn_wmma_f32_16x16x32_f16(
                          false, a0[jt], false, b0, (short)0, vzero, false, false);
#pragma unroll
        for (int jt = 0; jt < 4; ++jt)
            acc[jt] = __builtin_amdgcn_wmma_f32_16x16x32_f16(
                          false, a1[jt], false, b1, (short)0, acc[jt], false, false);
#pragma unroll
        for (int jt = 0; jt < 4; ++jt) {
            v8h hv;
#pragma unroll
            for (int v = 0; v < 8; ++v) hv[v] = (_Float16)acc[jt][v];
            // T[m][j], m = mt*16+ln, j = jt*16 + 8g + v  (8 contiguous halves)
            *(v8h*)&sT[(mt * 16 + ln) * PKD + jt * 16 + 8 * g] = hv;
        }
    }
    __syncthreads();

    // ============ GEMM2: roi[96x96] = T[96x64] @ M^T[64x96] + argmax =============
    float bestV = -3.4e38f;
    int   bestP = 0x7fffffff;
    {
        const int mi = wid;
        v16h a0 = load_afrag(&sT[(mi * 16 + ln) * PKD + 0 * 32 + 8 * g]);
        v16h a1 = load_afrag(&sT[(mi * 16 + ln) * PKD + 1 * 32 + 8 * g]);
        v8f acc[6];
#pragma unroll
        for (int nj = 0; nj < 6; ++nj) {
            v16h b0 = load_afrag(&sM[(nj * 16 + ln) * PKD + 0 * 32 + 8 * g]);
            acc[nj] = __builtin_amdgcn_wmma_f32_16x16x32_f16(
                          false, a0, false, b0, (short)0, vzero, false, false);
        }
#pragma unroll
        for (int nj = 0; nj < 6; ++nj) {
            v16h b1 = load_afrag(&sM[(nj * 16 + ln) * PKD + 1 * 32 + 8 * g]);
            acc[nj] = __builtin_amdgcn_wmma_f32_16x16x32_f16(
                          false, a1, false, b1, (short)0, acc[nj], false, false);
        }
#pragma unroll
        for (int nj = 0; nj < 6; ++nj) {
#pragma unroll
            for (int v = 0; v < 8; ++v) {
                int row = mi * 16 + v + 8 * g;
                int col = nj * 16 + ln;
                int p = row * ROI + col;
                float val = acc[nj][v];
                if (val > bestV || (val == bestV && p < bestP)) { bestV = val; bestP = p; }
            }
        }
    }

    // wave32 reduction (first-index tie-break)
#pragma unroll
    for (int off = 16; off > 0; off >>= 1) {
        float ov = __shfl_xor(bestV, off, 32);
        int   op = __shfl_xor(bestP, off, 32);
        if (ov > bestV || (ov == bestV && op < bestP)) { bestV = ov; bestP = op; }
    }
    if (lane == 0) { sRv[wid] = bestV; sRp[wid] = bestP; }
    __syncthreads();

    if (tid == 0) {
        float bv = sRv[0]; int bp = sRp[0];
#pragma unroll
        for (int w = 1; w < 6; ++w) {
            if (sRv[w] > bv || (sRv[w] == bv && sRp[w] < bp)) { bv = sRv[w]; bp = sRp[w]; }
        }
        int y = bp / ROI;
        int x = bp - y * ROI;

        // exact fp32 rescore at (y, x): 4x4 bicubic taps against fp32 mask
        int y0, x0; float wy[4], wx[4];
        bicubic_taps(y, &y0, wy);
        bicubic_taps(x, &x0, wx);
        float score = 0.0f;
#pragma unroll
        for (int a = 0; a < 4; ++a) {
            int iy = y0 + a;
            if (iy >= HIN) continue;
            float rowacc = 0.0f;
#pragma unroll
            for (int b = 0; b < 4; ++b) {
                int jx = x0 + b;
                if (jx >= HIN) continue;
                rowacc += wx[b] * m[iy * HIN + jx];
            }
            score += wy[a] * rowacc;
        }

        int r = rk / KPT;
        int k = rk - r * KPT;
        const float* bx = boxes + r * 4;
        float len0 = fmaxf(bx[2] - bx[0], 1.0f);
        float len1 = fmaxf(bx[3] - bx[1], 1.0f);
        float p0 = ((float)y + 0.5f) * (len0 / (float)ROI) + bx[0];
        float p1 = ((float)x + 0.5f) * (len1 / (float)ROI) + bx[1];

        // predictions [R,3,K] then scores [R,K]
        out[r * 3 * KPT + 0 * KPT + k] = p0;
        out[r * 3 * KPT + 1 * KPT + k] = p1;
        out[r * 3 * KPT + 2 * KPT + k] = 1.0f;
        out[RR * 3 * KPT + rk] = score;
    }
}

extern "C" void kernel_launch(void* const* d_in, const int* in_sizes, int n_in,
                              void* d_out, int out_size, void* d_ws, size_t ws_size,
                              hipStream_t stream) {
    (void)in_sizes; (void)n_in; (void)d_ws; (void)ws_size; (void)out_size;
    const float* masks = (const float*)d_in[0];   // [512,17,56,56] f32
    const float* boxes = (const float*)d_in[1];   // [512,4] f32
    float* out = (float*)d_out;                   // [512*3*17] preds + [512*17] scores
    kp_heatmap_kernel<<<RR * KPT, NTHR, 0, stream>>>(masks, boxes, out);
}
